// BDHLayer_21191368638897
// MI455X (gfx1250) — matrix operand order repaired and checked by reading, not compiled, not run on hardware
//
#include <hip/hip_runtime.h>
#include <hip/hip_bf16.h>

// ---------------------------------------------------------------------------
// BDH layer for MI455X (gfx1250): all GEMMs via v_wmma_f32_16x16x32_f16.
// ---------------------------------------------------------------------------

typedef __attribute__((ext_vector_type(16))) _Float16 v16h;
typedef __attribute__((ext_vector_type(8)))  _Float16 v8h;
typedef __attribute__((ext_vector_type(4)))  _Float16 v4h;
typedef __attribute__((ext_vector_type(8)))  float    v8f;

#define D_    256
#define NH_   4
#define N_    4096
#define NHN_  16384
#define BT_   2048      // B*T rows
#define T_    1024
#define CH_   128       // GLA chunk length
#define NCH_  16        // total chunks over B*T
#define EPS_  1e-5f

__device__ __forceinline__ v8f wmma16(v16h a, v16h b, v8f c) {
  return __builtin_amdgcn_wmma_f32_16x16x32_f16(false, a, false, b, (short)0, c,
                                                false, false);
}

// A-operand fragment (16x32 f16, MxK): lane holds row (lane&15); K halves per
// ISA layout: vgprs 0..3 = K kh..kh+7, vgprs 4..7 = K 16+kh..16+kh+7, kh=(lane>>4)*8.
__device__ __forceinline__ v16h frag_a(const _Float16* lds, int row, int ldk,
                                       int kbase, int lane) {
  int kh = (lane >> 4) << 3;
  v8h lo = *(const v8h*)(lds + row * ldk + kbase + kh);
  v8h hi = *(const v8h*)(lds + row * ldk + kbase + 16 + kh);
  return __builtin_shufflevector(lo, hi, 0,1,2,3,4,5,6,7,8,9,10,11,12,13,14,15);
}

// B-operand fragment (32x16 f16, KxN) staged as [N][K]: lane holds col (lane&15),
// K kb..kb+7 then 16+kb..16+kb+7, kb=(lane>>4)*8.
__device__ __forceinline__ v16h frag_b(const _Float16* lds, int nrow, int ldk,
                                       int kbase, int lane) {
  int kb = (lane >> 4) << 3;
  v8h lo = *(const v8h*)(lds + nrow * ldk + kbase + kb);
  v8h hi = *(const v8h*)(lds + nrow * ldk + kbase + 16 + kb);
  return __builtin_shufflevector(lo, hi, 0,1,2,3,4,5,6,7,8,9,10,11,12,13,14,15);
}

// ----------------------------- prep kernels --------------------------------

__global__ __launch_bounds__(256) void k_f32_to_f16(const float* __restrict__ s,
                                                    _Float16* __restrict__ d,
                                                    long n) {
  long i = (long)blockIdx.x * 256 + threadIdx.x;
  if (i < n) d[i] = (_Float16)s[i];
}

// dst[c][r] = (f16) src[r][c], batched over blockIdx.z (stride R*C both sides)
__global__ void k_transpose_f16(const float* __restrict__ src,
                                _Float16* __restrict__ dst, int R, int C) {
  long bz = blockIdx.z;
  src += bz * (long)R * C;
  dst += bz * (long)R * C;
  int c = blockIdx.x * 16 + threadIdx.x;
  int r = blockIdx.y * 16 + threadIdx.y;
  if (r < R && c < C) dst[(long)c * R + r] = (_Float16)src[(long)r * C + c];
}

// --------------------- K1: e/g = x @ {W_enc, W_gate} -----------------------
// M=2048 N=16384 K=256. WG tile 32x256, 8 waves of 16x64, dual accumulators.
__global__ __launch_bounds__(256) void k_encode(const _Float16* __restrict__ x16,
                                                const _Float16* __restrict__ We,
                                                const _Float16* __restrict__ Wg,
                                                _Float16* __restrict__ xs,
                                                _Float16* __restrict__ gate) {
  __shared__ __align__(16) _Float16 Ast[32 * 32];
  __shared__ __align__(16) _Float16 Be[256 * 32];
  __shared__ __align__(16) _Float16 Bg[256 * 32];
  int tid = threadIdx.x, lane = tid & 31, wv = tid >> 5;
  int t0 = blockIdx.y * 32;
  long n0 = (long)blockIdx.x * 256;
  int mw = wv & 1, nw = wv >> 1;
  v8f ae[4] = {}, ag[4] = {};
  for (int kk = 0; kk < 256; kk += 32) {
    {
      int r = tid >> 3, c4 = (tid & 7) * 4;
      *(v4h*)(Ast + r * 32 + c4) =
          *(const v4h*)(x16 + (long)(t0 + r) * 256 + kk + c4);
    }
    {
      int n = tid;
      const _Float16* pe = We + (n0 + n) * 256 + kk;
      const _Float16* pg = Wg + (n0 + n) * 256 + kk;
      __builtin_prefetch(pe + 32, 0, 1);
      __builtin_prefetch(pg + 32, 0, 1);
#pragma unroll
      for (int c = 0; c < 32; c += 8) {
        *(v8h*)(Be + n * 32 + c) = *(const v8h*)(pe + c);
        *(v8h*)(Bg + n * 32 + c) = *(const v8h*)(pg + c);
      }
    }
    __syncthreads();
    v16h a = frag_a(Ast, mw * 16 + (lane & 15), 32, 0, lane);
#pragma unroll
    for (int s = 0; s < 4; ++s) {
      v16h be = frag_b(Be, nw * 64 + s * 16 + (lane & 15), 32, 0, lane);
      ae[s] = wmma16(a, be, ae[s]);
      v16h bg = frag_b(Bg, nw * 64 + s * 16 + (lane & 15), 32, 0, lane);
      ag[s] = wmma16(a, bg, ag[s]);
    }
    __syncthreads();
  }
  int hi = lane >> 4, nl = lane & 15;
#pragma unroll
  for (int s = 0; s < 4; ++s)
#pragma unroll
    for (int r = 0; r < 8; ++r) {
      int ml = mw * 16 + hi * 8 + r;
      long n = n0 + nw * 64 + s * 16 + nl;
      long gi = (long)(t0 + ml) * NHN_ + n;
      float e = ae[s][r];
      float se = fmaxf(e, 0.0f); se *= se;
      xs[gi] = (_Float16)se;
      float g = ag[s][r];
      float sg = fmaxf(g, 0.0f); sg = sg * sg * (1.0f / 1024.0f);
      gate[gi] = (_Float16)sg;
    }
}

// ---------- K1.5: per-(chunk, feature) cumsum + RoPE + decay prebake -------
__global__ __launch_bounds__(256) void k_gates(const _Float16* __restrict__ xs,
                                               const _Float16* __restrict__ gate,
                                               _Float16* __restrict__ qg,
                                               _Float16* __restrict__ kg,
                                               _Float16* __restrict__ kgdbT,
                                               float* __restrict__ db) {
  int n = blockIdx.x * 256 + threadIdx.x;   // 0..16383
  int tc = blockIdx.y;                      // global chunk 0..15
  int c = n & 255;                          // index within CS=256 rope block
  int fi = c & 127;
  bool hiH = c >= 128;
  long p_lo = hiH ? n - 128 : n;
  long p_hi = hiH ? n : n + 128;
  float inv_freq = exp2f(-0.140625f * (float)fi);   // ROPE_BASE^(-2fi/256)
  float lscl = log2f((2.0f * fi + 102.4f) / 358.4f);
  const float qscale = 0.015625f;                   // N^-0.5
  float bsum = 0.0f;
  for (int j = 0; j < 128; ++j) {
    long t = (long)tc * 128 + j;
    int tloc = (int)(t & (T_ - 1));
    bsum -= (float)gate[t * NHN_ + n];
    float s1 = (float)xs[t * NHN_ + p_lo];
    float s2 = (float)xs[t * NHN_ + p_hi];
    float fr = (float)tloc * inv_freq;
    float sc = exp2f(((float)(tloc - 512) * (1.0f / 512.0f)) * lscl);
    float cv = cosf(fr) * sc, sv = sinf(fr) * sc;
    float xr = hiH ? (s1 * sv + s2 * cv) : (s1 * cv - s2 * sv);
    float eb = expf(bsum);
    qg[t * NHN_ + n] = (_Float16)(xr * qscale * eb);
    kg[t * NHN_ + n] = (_Float16)(xr / eb);
  }
  float dbv = expf(bsum);
  db[(long)tc * NHN_ + n] = dbv;
  _Float16* kt = kgdbT + ((long)tc * NHN_ + n) * 128;
  for (int j = 0; j < 128; ++j) {
    long t = (long)tc * 128 + j;
    kt[j] = (_Float16)((float)kg[t * NHN_ + n] * dbv);   // kg*db, [k][j] layout
  }
}

// ------------------- K2a: o = mask(qg kg^T) v + qg h^T ---------------------
// One chunk step; grid (itile=4, bh=8). WG: 32 rows x 256 vd.
__global__ __launch_bounds__(256) void k_gla_o(const _Float16* __restrict__ qg,
                                               const _Float16* __restrict__ kg,
                                               const _Float16* __restrict__ vT,
                                               const float* __restrict__ hT,
                                               float* __restrict__ yKV,
                                               int cstep) {
  __shared__ __align__(16) _Float16 Asa[32 * 32];
  __shared__ __align__(16) _Float16 Bst[256 * 32];
  __shared__ __align__(16) _Float16 Amat[32 * 128];
  int tid = threadIdx.x, lane = tid & 31, wv = tid >> 5;
  int itile = blockIdx.x, bh = blockIdx.y;
  int b = bh >> 2, h = bh & 3;
  int tchunk = b * 8 + cstep;
  int tC = tchunk * 128;
  int t0 = tC + itile * 32;
  long nbase = (long)h * N_;
  // ---- phase 1: A(32x128) = qg . kg^T over K=4096, wave tile 16x32 ----
  int mw = wv & 1, nw = wv >> 1;
  v8f accA[2] = {};
  for (int kk = 0; kk < N_; kk += 32) {
    {
      int r = tid >> 3, c4 = (tid & 7) * 4;
      *(v4h*)(Asa + r * 32 + c4) =
          *(const v4h*)(qg + (long)(t0 + r) * NHN_ + nbase + kk + c4);
    }
    {
      int jn = tid >> 1, c16 = (tid & 1) * 16;
      const _Float16* gp = kg + (long)(tC + jn) * NHN_ + nbase + kk + c16;
      *(v8h*)(Bst + jn * 32 + c16) = *(const v8h*)gp;
      *(v8h*)(Bst + jn * 32 + c16 + 8) = *(const v8h*)(gp + 8);
    }
    __syncthreads();
    v16h a = frag_a(Asa, mw * 16 + (lane & 15), 32, 0, lane);
#pragma unroll
    for (int s = 0; s < 2; ++s) {
      v16h bf = frag_b(Bst, nw * 32 + s * 16 + (lane & 15), 32, 0, lane);
      accA[s] = wmma16(a, bf, accA[s]);
    }
    __syncthreads();
  }
  {
    int hi = lane >> 4, nl = lane & 15;
#pragma unroll
    for (int s = 0; s < 2; ++s)
#pragma unroll
      for (int r = 0; r < 8; ++r) {
        int ml = mw * 16 + hi * 8 + r;          // 0..31
        int j = nw * 32 + s * 16 + nl;          // 0..127
        int ipos = itile * 32 + ml;             // causal mask (tril)
        float v = (j <= ipos) ? accA[s][r] : 0.0f;
        Amat[ml * 128 + j] = (_Float16)v;
      }
  }
  __syncthreads();
  // ---- phase 2: o(32x256), wave tile 16x64 ----
  int mw2 = wv & 1, nw2 = wv >> 1;
  v8f acc[4] = {};
  const _Float16* vTc = vT + (long)tchunk * 256 * 128;
  for (int kj = 0; kj < 128; kj += 32) {        // intra-chunk A @ v
    {
      int n = tid;
#pragma unroll
      for (int c = 0; c < 32; c += 8)
        *(v8h*)(Bst + n * 32 + c) = *(const v8h*)(vTc + (long)n * 128 + kj + c);
    }
    __syncthreads();
    v16h a = frag_a(Amat, mw2 * 16 + (lane & 15), 128, kj, lane);
#pragma unroll
    for (int s = 0; s < 4; ++s) {
      v16h bf = frag_b(Bst, nw2 * 64 + s * 16 + (lane & 15), 32, 0, lane);
      acc[s] = wmma16(a, bf, acc[s]);
    }
    __syncthreads();
  }
  const float* hTb = hT + (long)bh * 256 * N_;  // qg @ h^T (state term)
  for (int kk = 0; kk < N_; kk += 32) {
    {
      int r = tid >> 3, c4 = (tid & 7) * 4;
      *(v4h*)(Asa + r * 32 + c4) =
          *(const v4h*)(qg + (long)(t0 + r) * NHN_ + nbase + kk + c4);
    }
    {
      int n = tid;
      const float* hp = hTb + (long)n * N_ + kk;
#pragma unroll
      for (int c = 0; c < 32; c += 4) {
        float4 f = *(const float4*)(hp + c);
        _Float16* q = Bst + n * 32 + c;
        q[0] = (_Float16)f.x; q[1] = (_Float16)f.y;
        q[2] = (_Float16)f.z; q[3] = (_Float16)f.w;
      }
    }
    __syncthreads();
    v16h a = frag_a(Asa, mw2 * 16 + (lane & 15), 32, 0, lane);
#pragma unroll
    for (int s = 0; s < 4; ++s) {
      v16h bf = frag_b(Bst, nw2 * 64 + s * 16 + (lane & 15), 32, 0, lane);
      acc[s] = wmma16(a, bf, acc[s]);
    }
    __syncthreads();
  }
  int hi = lane >> 4, nl = lane & 15;
#pragma unroll
  for (int s = 0; s < 4; ++s)
#pragma unroll
    for (int r = 0; r < 8; ++r) {
      int ml = mw2 * 16 + hi * 8 + r;
      int col = nw2 * 64 + s * 16 + nl;
      yKV[(long)(t0 + ml) * 1024 + h * 256 + col] = acc[s][r];
    }
}

// ------------- K2b: hT = hT*db + vT @ (kg*db)   (per chunk) ----------------
// grid (ktile=64, bh=8). WG tile: 256(vd) x 64(k); wave 64x32 (8 accums).
__global__ __launch_bounds__(256) void k_gla_state(const _Float16* __restrict__ vT,
                                                   const _Float16* __restrict__ kgdbT,
                                                   const float* __restrict__ db,
                                                   float* __restrict__ hT,
                                                   int cstep) {
  __shared__ __align__(16) _Float16 Ast[256 * 32];
  __shared__ __align__(16) _Float16 Bst[64 * 32];
  int tid = threadIdx.x, lane = tid & 31, wv = tid >> 5;
  int ktile = blockIdx.x, bh = blockIdx.y;
  int b = bh >> 2, h = bh & 3;
  int tchunk = b * 8 + cstep;
  const _Float16* vTc = vT + (long)tchunk * 256 * 128;
  long nbase = (long)h * N_;
  int k0 = ktile * 64;
  const _Float16* kbT = kgdbT + ((long)tchunk * NHN_ + nbase + k0) * 128;
  int mw = wv & 3, nw = wv >> 2;
  v8f acc[4][2] = {};
  for (int kj = 0; kj < 128; kj += 32) {
    {
      int n = tid;
#pragma unroll
      for (int c = 0; c < 32; c += 8)
        *(v8h*)(Ast + n * 32 + c) = *(const v8h*)(vTc + (long)n * 128 + kj + c);
    }
    {
      int rr = tid >> 2, c8 = (tid & 3) * 8;
      *(v8h*)(Bst + rr * 32 + c8) = *(const v8h*)(kbT + (long)rr * 128 + kj + c8);
    }
    __syncthreads();
    v16h av[4];
#pragma unroll
    for (int sa = 0; sa < 4; ++sa)
      av[sa] = frag_a(Ast, mw * 64 + sa * 16 + (lane & 15), 32, 0, lane);
#pragma unroll
    for (int sb = 0; sb < 2; ++sb) {
      v16h bf = frag_b(Bst, nw * 32 + sb * 16 + (lane & 15), 32, 0, lane);
#pragma unroll
      for (int sa = 0; sa < 4; ++sa) acc[sa][sb] = wmma16(av[sa], bf, acc[sa][sb]);
    }
    __syncthreads();
  }
  float* hTb = hT + (long)bh * 256 * N_;
  int hi = lane >> 4, nl = lane & 15;
#pragma unroll
  for (int sa = 0; sa < 4; ++sa)
#pragma unroll
    for (int sb = 0; sb < 2; ++sb)
#pragma unroll
      for (int r = 0; r < 8; ++r) {
        int vd = mw * 64 + sa * 16 + hi * 8 + r;
        int kgl = k0 + nw * 32 + sb * 16 + nl;
        float dv = db[(long)tchunk * NHN_ + nbase + kgl];
        long idx = (long)vd * N_ + kgl;
        hTb[idx] = hTb[idx] * dv + acc[sa][sb][r];
      }
}

// ---------------------- K3a: layernorm rows of yKV -------------------------
__global__ __launch_bounds__(256) void k_ln_ykv(const float* __restrict__ yKV,
                                                _Float16* __restrict__ yn) {
  __shared__ float red[256];
  int tid = threadIdx.x;
  long row = blockIdx.x;                      // t*4 + h
  float v = yKV[row * 256 + tid];
  red[tid] = v;
  __syncthreads();
  for (int s = 128; s > 0; s >>= 1) {
    if (tid < s) red[tid] += red[tid + s];
    __syncthreads();
  }
  float mean = red[0] * (1.0f / 256.0f);
  __syncthreads();
  float dvv = v - mean;
  red[tid] = dvv * dvv;
  __syncthreads();
  for (int s = 128; s > 0; s >>= 1) {
    if (tid < s) red[tid] += red[tid + s];
    __syncthreads();
  }
  float var = red[0] * (1.0f / 256.0f);
  yn[row * 256 + tid] = (_Float16)(dvv * rsqrtf(var + EPS_));
}

// ------ K3b: xy = xs * sqrelu(yn @ W_encv[h]) per head (M2048 N4096 K256) --
__global__ __launch_bounds__(256) void k_decode_v(const _Float16* __restrict__ yn,
                                                  const _Float16* __restrict__ Wv,
                                                  const _Float16* __restrict__ xs,
                                                  _Float16* __restrict__ xy) {
  __shared__ __align__(16) _Float16 Ast[32 * 32];
  __shared__ __align__(16) _Float16 Bst[128 * 32];
  int tid = threadIdx.x, lane = tid & 31, wv = tid >> 5;
  int n0 = blockIdx.x * 128, t0 = blockIdx.y * 32, h = blockIdx.z;
  int mw = wv & 1, nw = wv >> 1;
  v8f acc[2] = {};
  for (int kk = 0; kk < 256; kk += 32) {
    {
      int r = tid >> 3, c4 = (tid & 7) * 4;
      *(v4h*)(Ast + r * 32 + c4) =
          *(const v4h*)(yn + (long)(t0 + r) * 1024 + h * 256 + kk + c4);
    }
    {
      int rr = tid >> 1, c16 = (tid & 1) * 16;
      const _Float16* wp = Wv + ((long)h * N_ + n0 + rr) * 256 + kk + c16;
      *(v8h*)(Bst + rr * 32 + c16) = *(const v8h*)wp;
      *(v8h*)(Bst + rr * 32 + c16 + 8) = *(const v8h*)(wp + 8);
    }
    __syncthreads();
    v16h a = frag_a(Ast, mw * 16 + (lane & 15), 32, 0, lane);
#pragma unroll
    for (int s = 0; s < 2; ++s) {
      v16h bf = frag_b(Bst, nw * 32 + s * 16 + (lane & 15), 32, 0, lane);
      acc[s] = wmma16(a, bf, acc[s]);
    }
    __syncthreads();
  }
  int hi = lane >> 4, nl = lane & 15;
#pragma unroll
  for (int s = 0; s < 2; ++s)
#pragma unroll
    for (int r = 0; r < 8; ++r) {
      int ml = mw * 16 + hi * 8 + r;
      int col = nw * 32 + s * 16 + nl;
      float ys = fmaxf(acc[s][r], 0.0f); ys *= ys;
      long gi = (long)(t0 + ml) * NHN_ + (long)h * N_ + n0 + col;
      xy[gi] = (_Float16)(ys * (float)xs[gi]);
    }
}

// --- K4: out = rmsnorm(layernorm(xy @ W_dec) + x)  (M2048 N256 K16384) -----
__global__ __launch_bounds__(256) void k_final(const _Float16* __restrict__ xy,
                                               const _Float16* __restrict__ Wd,
                                               const float* __restrict__ x,
                                               float* __restrict__ out) {
  __shared__ __align__(16) _Float16 Ast[32 * 32];
  __shared__ __align__(16) _Float16 Bst[256 * 32];
  __shared__ float Y[32 * 256];
  __shared__ float stats[64];
  int tid = threadIdx.x, lane = tid & 31, wv = tid >> 5;
  int t0 = blockIdx.x * 32;
  int mw = wv & 1, nw = wv >> 1;
  v8f acc[4] = {};
  for (int kk = 0; kk < NHN_; kk += 32) {
    {
      int r = tid >> 3, c4 = (tid & 7) * 4;
      *(v4h*)(Ast + r * 32 + c4) =
          *(const v4h*)(xy + (long)(t0 + r) * NHN_ + kk + c4);
    }
    {
      int d = tid;
      const _Float16* wp = Wd + (long)d * NHN_ + kk;
      __builtin_prefetch(wp + 32, 0, 1);
#pragma unroll
      for (int c = 0; c < 32; c += 8)
        *(v8h*)(Bst + d * 32 + c) = *(const v8h*)(wp + c);
    }
    __syncthreads();
    v16h a = frag_a(Ast, mw * 16 + (lane & 15), 32, 0, lane);
#pragma unroll
    for (int s = 0; s < 4; ++s) {
      v16h bf = frag_b(Bst, nw * 64 + s * 16 + (lane & 15), 32, 0, lane);
      acc[s] = wmma16(a, bf, acc[s]);
    }
    __syncthreads();
  }
  {
    int hi = lane >> 4, nl = lane & 15;
#pragma unroll
    for (int s = 0; s < 4; ++s)
#pragma unroll
      for (int r = 0; r < 8; ++r) {
        int ml = mw * 16 + hi * 8 + r;
        int col = nw * 64 + s * 16 + nl;
        Y[ml * 256 + col] = acc[s][r];
      }
  }
  __syncthreads();
  if (tid < 32) {                    // layer_norm stats per row
    float sum = 0.0f;
    for (int d = 0; d < 256; ++d) sum += Y[tid * 256 + d];
    float mean = sum * (1.0f / 256.0f);
    float vs = 0.0f;
    for (int d = 0; d < 256; ++d) {
      float dd = Y[tid * 256 + d] - mean;
      vs += dd * dd;
    }
    stats[tid * 2] = mean;
    stats[tid * 2 + 1] = rsqrtf(vs * (1.0f / 256.0f) + EPS_);
  }
  __syncthreads();
  for (int r = 0; r < 32; ++r) {     // z = ln(yMLP) + x
    float z = (Y[r * 256 + tid] - stats[r * 2]) * stats[r * 2 + 1] +
              x[(long)(t0 + r) * 256 + tid];
    Y[r * 256 + tid] = z;
  }
  __syncthreads();
  if (tid < 32) {                    // rms stats per row
    float ss = 0.0f;
    for (int d = 0; d < 256; ++d) ss += Y[tid * 256 + d] * Y[tid * 256 + d];
    stats[tid * 2] = rsqrtf(ss * (1.0f / 256.0f) + EPS_);
  }
  __syncthreads();
  for (int r = 0; r < 32; ++r)
    out[(long)(t0 + r) * 256 + tid] = Y[r * 256 + tid] * stats[r * 2];
}

// ------------------------------- host side ---------------------------------

extern "C" void kernel_launch(void* const* d_in, const int* in_sizes, int n_in,
                              void* d_out, int out_size, void* d_ws,
                              size_t ws_size, hipStream_t stream) {
  const float* x     = (const float*)d_in[0];   // (2048, 256)
  const float* W_enc = (const float*)d_in[1];   // (256, 16384)
  const float* W_gate= (const float*)d_in[2];   // (256, 16384)
  const float* W_dec = (const float*)d_in[3];   // (16384, 256)
  const float* W_encv= (const float*)d_in[4];   // (4, 256, 4096)
  float* out = (float*)d_out;

  char* w = (char*)d_ws;
  size_t cur = 0;
  auto alloc = [&](size_t bytes) {
    void* p = w + cur;
    cur += (bytes + 255) & ~(size_t)255;
    return p;
  };
  _Float16* x16    = (_Float16*)alloc((size_t)BT_ * D_ * 2);
  _Float16* Wt_enc = (_Float16*)alloc((size_t)NHN_ * D_ * 2);
  _Float16* Wt_gate= (_Float16*)alloc((size_t)NHN_ * D_ * 2);
  _Float16* Wt_encv= (_Float16*)alloc((size_t)NH_ * N_ * D_ * 2);
  _Float16* Wt_dec = (_Float16*)alloc((size_t)D_ * NHN_ * 2);
  _Float16* xs16   = (_Float16*)alloc((size_t)BT_ * NHN_ * 2);
  _Float16* gate16 = (_Float16*)alloc((size_t)BT_ * NHN_ * 2);
  _Float16* qg16   = (_Float16*)alloc((size_t)BT_ * NHN_ * 2);  // reused as xy
  _Float16* kg16   = (_Float16*)alloc((size_t)BT_ * NHN_ * 2);
  _Float16* kgdbT  = (_Float16*)alloc((size_t)NCH_ * NHN_ * CH_ * 2);
  _Float16* vT16   = (_Float16*)alloc((size_t)NCH_ * D_ * CH_ * 2);
  float*    db     = (float*)alloc((size_t)NCH_ * NHN_ * 4);
  float*    hT     = (float*)alloc((size_t)8 * D_ * N_ * 4);    // B*H states
  float*    yKV    = (float*)alloc((size_t)BT_ * NH_ * D_ * 4);
  _Float16* yn16   = (_Float16*)alloc((size_t)BT_ * NH_ * D_ * 2);
  _Float16* xy16   = qg16;   // alias: qg dead once GLA chunk loop finished

  // prep: f16 conversions / transposes into [N][K] layouts
  k_f32_to_f16<<<(BT_ * D_) / 256, 256, 0, stream>>>(x, x16, (long)BT_ * D_);
  k_transpose_f16<<<dim3(NHN_ / 16, D_ / 16, 1), dim3(16, 16), 0, stream>>>(
      W_enc, Wt_enc, D_, NHN_);
  k_transpose_f16<<<dim3(NHN_ / 16, D_ / 16, 1), dim3(16, 16), 0, stream>>>(
      W_gate, Wt_gate, D_, NHN_);
  k_transpose_f16<<<dim3(N_ / 16, D_ / 16, NH_), dim3(16, 16), 0, stream>>>(
      W_encv, Wt_encv, D_, N_);
  k_transpose_f16<<<dim3(D_ / 16, NHN_ / 16, 1), dim3(16, 16), 0, stream>>>(
      W_dec, Wt_dec, NHN_, D_);
  // per-chunk transposed V tiles: vT[chunk][d][j] = x[chunk*128+j][d]
  k_transpose_f16<<<dim3(D_ / 16, CH_ / 16, NCH_), dim3(16, 16), 0, stream>>>(
      x, vT16, CH_, D_);

  // encoder GEMMs + sqrelu
  k_encode<<<dim3(NHN_ / 256, BT_ / 32), 256, 0, stream>>>(x16, Wt_enc, Wt_gate,
                                                           xs16, gate16);
  // decay cumsum + RoPE prebake
  k_gates<<<dim3(NHN_ / 256, NCH_), 256, 0, stream>>>(xs16, gate16, qg16, kg16,
                                                      kgdbT, db);
  // GLA recurrence over 8 chunk steps (both batches in parallel)
  hipMemsetAsync(hT, 0, (size_t)8 * D_ * N_ * 4, stream);
  for (int c = 0; c < 8; ++c) {
    k_gla_o<<<dim3(4, 8), 256, 0, stream>>>(qg16, kg16, vT16, hT, yKV, c);
    k_gla_state<<<dim3(64, 8), 256, 0, stream>>>(vT16, kgdbT, db, hT, c);
  }
  // layernorm + value decoder + gating
  k_ln_ykv<<<BT_ * NH_, 256, 0, stream>>>(yKV, yn16);
  k_decode_v<<<dim3(N_ / 128, BT_ / 32, NH_), 256, 0, stream>>>(yn16, Wt_encv,
                                                                xs16, xy16);
  // final projection + layernorm + residual + rmsnorm
  k_final<<<BT_ / 32, 256, 0, stream>>>(xy16, Wt_dec, x, out);
  (void)in_sizes; (void)n_in; (void)out_size; (void)ws_size;
}
